// custom_rbln_paged_attn_prefill_11012296147684
// MI455X (gfx1250) — matrix operand order repaired and checked by reading, not compiled, not run on hardware
//
#include <hip/hip_runtime.h>

// Paged-attention prefill for MI455X (gfx1250): flash-attention with
// bf16 V_WMMA_F32_16X16X32_BF16 for both QK^T and PV, f32 accumulation.
// 64 keys per iteration (32 WMMAs) to amortize softmax VALU overhead;
// V staged transposed in LDS; log2-domain softmax via raw v_exp_f32.

#define NKV     8
#define NGRP    4
#define QLEN    2048
#define HD      128
#define MAXSEQ  4096
#define BM      128     // q rows per workgroup (8 waves x 16 rows)
#define BN      64      // keys per inner iteration
#define KSTRIDE 136     // K LDS row stride (bf16 elems), padded vs 128
#define VSTRIDE 72      // Vt LDS row stride (bf16 elems), padded vs 64
#define PSTRIDE 72      // per-wave P scratch row stride (bf16 elems)

typedef __attribute__((ext_vector_type(16))) __bf16 v16bf;
typedef __attribute__((ext_vector_type(8)))  float  v8f;

#define LOG2E 1.4426950408889634f

union V16U {
  v16bf          v;
  unsigned short u[16];
  unsigned int   w[8];
};

__device__ __forceinline__ unsigned short bf16u(float f) {
  __bf16 h = (__bf16)f;                       // RNE, native cvt on gfx1250
  return __builtin_bit_cast(unsigned short, h);
}
__device__ __forceinline__ unsigned int pack2(float a, float b) {
  return (unsigned int)bf16u(a) | ((unsigned int)bf16u(b) << 16);
}
// 32 contiguous bytes (16 bf16) from LDS -> fragment (two ds_load_b128)
__device__ __forceinline__ void ld16h(V16U& d, const unsigned short* p) {
  const uint4* q = (const uint4*)p;
  uint4 a = q[0], b = q[1];
  d.w[0] = a.x; d.w[1] = a.y; d.w[2] = a.z; d.w[3] = a.w;
  d.w[4] = b.x; d.w[5] = b.y; d.w[6] = b.z; d.w[7] = b.w;
}

__global__ __launch_bounds__(256, 1)
void paged_attn_prefill_wmma_gfx1250(
    const float* __restrict__ q, const float* __restrict__ k,
    const float* __restrict__ v, const float* __restrict__ mask,
    const float* __restrict__ kcache, const float* __restrict__ vcache,
    const int* __restrict__ seqp, const int* __restrict__ btab,
    const int* __restrict__ bszp, float* __restrict__ out)
{
  __shared__ unsigned short Ks[BN][KSTRIDE];   // key-major K tile
  __shared__ unsigned short Vt[HD][VSTRIDE];   // d-major (transposed) V tile
  __shared__ unsigned short Ps[8][16][PSTRIDE];// per-wave P transpose scratch

  const int tid    = threadIdx.x;
  const int wave   = tid >> 5;
  const int lane   = tid & 31;
  const int nlane  = lane & 15;
  const int laneHi = lane >> 4;          // 0: lanes 0-15, 1: lanes 16-31

  const int q0  = blockIdx.x * BM;
  const int g   = blockIdx.y;
  const int h   = blockIdx.z;
  const int seq = seqp[0];
  const int bs  = bszp[0];

  const float sc2 = 0.08838834764831845f * LOG2E;  // (1/sqrt(128)) * log2(e)

  // ---- Q fragments, A-layout with contiguous k = 16*laneHi + s for 16x32 tiles (4 chunks of K=32) ----
  const int qrow = q0 + wave * 16 + nlane;
  const float* qp = q + ((size_t)(h * NGRP + g) * QLEN + qrow) * HD;
  V16U aq[4];
  #pragma unroll
  for (int c = 0; c < 4; ++c) {
    const float* b0 = qp + c * 32 + laneHi * 16;
    #pragma unroll
    for (int i = 0; i < 4; ++i) {
      float2 f0 = *(const float2*)(b0 + 2 * i);
      float2 f1 = *(const float2*)(b0 + 8 + 2 * i);
      aq[c].w[i]     = pack2(f0.x, f0.y);
      aq[c].w[4 + i] = pack2(f1.x, f1.y);
    }
  }

  v8f o[8];
  #pragma unroll
  for (int t = 0; t < 8; ++t) o[t] = (v8f){0.f,0.f,0.f,0.f,0.f,0.f,0.f,0.f};
  float mrow[8], lrow[8];
  #pragma unroll
  for (int r = 0; r < 8; ++r) { mrow[r] = -3.0e38f; lrow[r] = 0.f; }

  const int nvis      = seq + q0 + BM;                 // keys [0, nvis)
  const int ntiles    = (nvis + BN - 1) / BN;
  const int qmax_wave = seq + q0 + wave * 16 + 15;     // causal frontier

  for (int kt = 0; kt < ntiles; ++kt) {
    const int j0 = kt * BN;

    __syncthreads();
    // ---- stage K (key-major) tile: 64 rows x 128 d ----
    for (int idx = tid; idx < BN * (HD / 4); idx += 256) {
      int row = idx >> 5;
      int cq  = (idx & 31) << 2;
      int j   = j0 + row;  if (j > nvis - 1) j = nvis - 1;   // clamp (masked anyway)
      const float* sk;
      if (j >= seq) {
        sk = k + ((size_t)h * QLEN + (j - seq)) * HD + cq;
      } else {
        int lb = j / bs, off = j - lb * bs, pb = btab[lb];
        sk = kcache + (((size_t)pb * NKV + h) * bs + off) * HD + cq;
      }
      float4 fk = *(const float4*)sk;
      *(uint2*)&Ks[row][cq] = make_uint2(pack2(fk.x, fk.y), pack2(fk.z, fk.w));
    }
    // ---- stage V transposed (d-major), 4x4 blocks -> ds_store_b64 ----
    for (int bidx = tid; bidx < (BN / 4) * (HD / 4); bidx += 256) {
      int krow = (bidx >> 5) << 2;           // key base (0..60)
      int cq   = (bidx & 31) << 2;           // d base
      float4 fr[4];
      #pragma unroll
      for (int rr = 0; rr < 4; ++rr) {
        int j = j0 + krow + rr;  if (j > nvis - 1) j = nvis - 1;
        const float* sv;
        if (j >= seq) {
          sv = v + ((size_t)h * QLEN + (j - seq)) * HD + cq;
        } else {
          int lb = j / bs, off = j - lb * bs, pb = btab[lb];
          sv = vcache + (((size_t)pb * NKV + h) * bs + off) * HD + cq;
        }
        fr[rr] = *(const float4*)sv;
      }
      *(uint2*)&Vt[cq + 0][krow] = make_uint2(pack2(fr[0].x, fr[1].x), pack2(fr[2].x, fr[3].x));
      *(uint2*)&Vt[cq + 1][krow] = make_uint2(pack2(fr[0].y, fr[1].y), pack2(fr[2].y, fr[3].y));
      *(uint2*)&Vt[cq + 2][krow] = make_uint2(pack2(fr[0].z, fr[1].z), pack2(fr[2].z, fr[3].z));
      *(uint2*)&Vt[cq + 3][krow] = make_uint2(pack2(fr[0].w, fr[1].w), pack2(fr[2].w, fr[3].w));
    }
    __syncthreads();

    if (j0 > qmax_wave) continue;    // fully masked for this wave (barriers uniform)

    // ---- S = Q * K^T : four 16x16 n-tiles, 4 K-chunks, double-buffered frags ----
    v8f s[4];
    #pragma unroll
    for (int t4 = 0; t4 < 4; ++t4) s[t4] = (v8f){0.f,0.f,0.f,0.f,0.f,0.f,0.f,0.f};
    V16U bk[2][4];
    #pragma unroll
    for (int t4 = 0; t4 < 4; ++t4)
      ld16h(bk[0][t4], &Ks[t4 * 16 + nlane][laneHi * 16]);
    #pragma unroll
    for (int c = 0; c < 4; ++c) {
      int cur = c & 1, nxt = cur ^ 1;
      if (c < 3) {
        #pragma unroll
        for (int t4 = 0; t4 < 4; ++t4)
          ld16h(bk[nxt][t4], &Ks[t4 * 16 + nlane][(c + 1) * 32 + laneHi * 16]);
      }
      #pragma unroll
      for (int t4 = 0; t4 < 4; ++t4)
        s[t4] = __builtin_amdgcn_wmma_f32_16x16x32_bf16(false, aq[c].v, false, bk[cur][t4].v,
                                                        (short)0, s[t4], false, false);
    }

    // ---- online softmax, log2 domain (rows m = r + 8*laneHi) ----
    #pragma unroll
    for (int r = 0; r < 8; ++r) {
      int m  = r + laneHi * 8;
      int qi = q0 + wave * 16 + m;
      const float* mp = mask + (size_t)qi * MAXSEQ;
      float a[4];
      #pragma unroll
      for (int t4 = 0; t4 < 4; ++t4) {
        int ja = j0 + nlane + 16 * t4;  if (ja > MAXSEQ - 1) ja = MAXSEQ - 1;
        a[t4] = s[t4][r] * sc2 + mp[ja] * LOG2E;
      }
      float mx = fmaxf(fmaxf(a[0], a[1]), fmaxf(a[2], a[3]));
      #pragma unroll
      for (int d = 8; d >= 1; d >>= 1) mx = fmaxf(mx, __shfl_xor(mx, d, 32));
      float mnew = fmaxf(mrow[r], mx);
      float corr = __builtin_amdgcn_exp2f(mrow[r] - mnew);
      float p[4], sum = 0.f;
      #pragma unroll
      for (int t4 = 0; t4 < 4; ++t4) { p[t4] = __builtin_amdgcn_exp2f(a[t4] - mnew); sum += p[t4]; }
      #pragma unroll
      for (int d = 8; d >= 1; d >>= 1) sum += __shfl_xor(sum, d, 32);
      mrow[r] = mnew;
      lrow[r] = lrow[r] * corr + sum;
      #pragma unroll
      for (int t = 0; t < 8; ++t) o[t][r] *= corr;
      #pragma unroll
      for (int t4 = 0; t4 < 4; ++t4)
        Ps[wave][m][nlane + 16 * t4] = bf16u(p[t4]);
    }
    // intra-wave LDS RAW fence (C-layout stores -> A-layout reloads)
    asm volatile("s_wait_dscnt 0" ::: "memory");

    // ---- P fragments (A-layout 16x64 = two 16x32 chunks) ----
    V16U ap0, ap1;
    ld16h(ap0, &Ps[wave][nlane][laneHi * 8]);       // note: halves 0..3 / 4..7 pattern
    ld16h(ap1, &Ps[wave][nlane][32 + laneHi * 8]);
    // ld16h reads 16 contiguous halves; this code's A order is [base..base+7] then [base+8..+15], base = laneHi*16 (matches Vt B loads)
    {
      const uint4* pa = (const uint4*)&Ps[wave][nlane][laneHi * 16];
      const uint4* pb = (const uint4*)&Ps[wave][nlane][8 + laneHi * 16];
      uint4 x = pa[0], y = pb[0];
      ap0.w[0] = x.x; ap0.w[1] = x.y; ap0.w[2] = x.z; ap0.w[3] = x.w;
      ap0.w[4] = y.x; ap0.w[5] = y.y; ap0.w[6] = y.z; ap0.w[7] = y.w;
      const uint4* pc = (const uint4*)&Ps[wave][nlane][32 + laneHi * 16];
      const uint4* pd = (const uint4*)&Ps[wave][nlane][40 + laneHi * 16];
      uint4 z = pc[0], w4 = pd[0];
      ap1.w[0] = z.x; ap1.w[1] = z.y; ap1.w[2] = z.z; ap1.w[3] = z.w;
      ap1.w[4] = w4.x; ap1.w[5] = w4.y; ap1.w[6] = w4.z; ap1.w[7] = w4.w;
    }

    // ---- O += P * V : 8 d-tiles x 2 k-chunks, double-buffered B fragments ----
    V16U bv[2][2];
    ld16h(bv[0][0], &Vt[nlane][laneHi * 16]);
    ld16h(bv[0][1], &Vt[nlane][32 + laneHi * 16]);
    #pragma unroll
    for (int t = 0; t < 8; ++t) {
      int cur = t & 1;
      if (t < 7) {
        ld16h(bv[cur ^ 1][0], &Vt[(t + 1) * 16 + nlane][laneHi * 16]);
        ld16h(bv[cur ^ 1][1], &Vt[(t + 1) * 16 + nlane][32 + laneHi * 16]);
      }
      o[t] = __builtin_amdgcn_wmma_f32_16x16x32_bf16(false, ap0.v, false, bv[cur][0].v,
                                                     (short)0, o[t], false, false);
      o[t] = __builtin_amdgcn_wmma_f32_16x16x32_bf16(false, ap1.v, false, bv[cur][1].v,
                                                     (short)0, o[t], false, false);
    }
  }

  // ---- epilogue: normalize and store ----
  #pragma unroll
  for (int r = 0; r < 8; ++r) {
    float inv = 1.0f / lrow[r];
    int m  = r + laneHi * 8;
    int qi = q0 + wave * 16 + m;
    size_t ob = ((size_t)(h * NGRP + g) * QLEN + qi) * HD;
    #pragma unroll
    for (int t = 0; t < 8; ++t)
      out[ob + t * 16 + nlane] = o[t][r] * inv;
  }
}

extern "C" void kernel_launch(void* const* d_in, const int* in_sizes, int n_in,
                              void* d_out, int out_size, void* d_ws, size_t ws_size,
                              hipStream_t stream) {
  const float* q      = (const float*)d_in[0];
  const float* k      = (const float*)d_in[1];
  const float* v      = (const float*)d_in[2];
  const float* mask   = (const float*)d_in[3];
  const float* kcache = (const float*)d_in[4];
  const float* vcache = (const float*)d_in[5];
  const int*   seq    = (const int*)d_in[6];
  const int*   btab   = (const int*)d_in[7];
  const int*   bsz    = (const int*)d_in[8];
  float*       out    = (float*)d_out;

  dim3 grid(QLEN / BM, NGRP, NKV);   // 16 x 4 x 8 = 512 workgroups
  dim3 block(256);
  hipLaunchKernelGGL(paged_attn_prefill_wmma_gfx1250, grid, block, 0, stream,
                     q, k, v, mask, kcache, vcache, seq, btab, bsz, out);
}
